// naive_attn_like_55757265437282
// MI455X (gfx1250) — compile-verified, matrix operand order checked by reading, not run on hardware
//
#include <hip/hip_runtime.h>
#include <stdint.h>

// ---------------------------------------------------------------------------
// Fused attention-like op for gfx1250 (MI455X), wave32 + WMMA f16 + async-LDS
//   x [16,2048,1024] f32 ; qk = x@Wqk^T+b ; v = x@Wv^T+b
//   out = softmax(qk @ qk^T) @ v   [16,2048,128] f32
// Phase 1: 512-thr blocks, LDS-staged f16 weights, 16 waves x 8 WMMA tiles,
//          x operand software-pipelined across k-steps.
// Phase 2: 256-thr blocks, flash-style online softmax, K/V double-buffered in
//          LDS via global_load_async_to_lds_b128 (+ s_wait_asynccnt),
//          2-deep pipelined LDS->WMMA operand streams.
// ---------------------------------------------------------------------------

typedef __attribute__((ext_vector_type(16))) _Float16 v16h;
typedef __attribute__((ext_vector_type(8)))  _Float16 v8h;
typedef __attribute__((ext_vector_type(8)))  float    v8f;
typedef __attribute__((ext_vector_type(4)))  float    v4f;

#define NB   16
#define NSEQ 2048
#define NDIM 1024
#define NDK  128

#define KPAD 136   // padded row (halves) for 32x128 K tile  (68 dwords/row)
#define VPAD 40    // padded row (halves) for 128x32 V tile / P / W tiles

static __device__ __forceinline__ v8f wmma_f16(v16h a, v16h b, v8f c) {
  return __builtin_amdgcn_wmma_f32_16x16x32_f16(false, a, false, b,
                                                (short)0, c, false, false);
}

// Assemble 16x32 f16 A/B operand from two contiguous 8-half (16B) runs.
// element i (i<8)  -> K = half*8 + i        (run at p)
// element i (i>=8) -> K = 16 + half*8 + i-8 (run at p+16)
static __device__ __forceinline__ v16h load_ab_f16(const _Float16* p) {
  v16h r;
  ((v8h*)&r)[0] = *(const v8h*)p;
  ((v8h*)&r)[1] = *(const v8h*)(p + 16);
  return r;
}

static __device__ __forceinline__ v16h load_ab_f32(const float* p) {
  v4f a0 = *(const v4f*)(p);
  v4f a1 = *(const v4f*)(p + 4);
  v4f a2 = *(const v4f*)(p + 16);
  v4f a3 = *(const v4f*)(p + 20);
  v16h r;
#pragma unroll
  for (int i = 0; i < 4; ++i) {
    r[i]      = (_Float16)a0[i];
    r[i + 4]  = (_Float16)a1[i];
    r[i + 8]  = (_Float16)a2[i];
    r[i + 12] = (_Float16)a3[i];
  }
  return r;
}

// ---- CDNA5 async global->LDS copy (ASYNCcnt path, ISA 08_async_tensor) ----
static __device__ __forceinline__ void async_b128(const _Float16* lds,
                                                  const _Float16* gaddr) {
  uint32_t ldsOff = (uint32_t)(uintptr_t)lds;   // flat addr[31:0] == LDS addr
  asm volatile("global_load_async_to_lds_b128 %0, %1, off"
               :: "v"(ldsOff), "v"(gaddr) : "memory");
}

#if __has_builtin(__builtin_amdgcn_s_wait_asynccnt)
#define WAIT_ASYNC(N) __builtin_amdgcn_s_wait_asynccnt(N)
#else
#define WAIT_ASYNC(N) asm volatile("s_wait_asynccnt " #N ::: "memory")
#endif

// ---------------------------------------------------------------------------
// Phase 1: fused QK / V projection.
// grid = 32768/128 blocks of 512 threads (16 waves).
// wave w: matrix = w>>3 (0:qk 1:v), token tile = w&7 (16 tokens each).
// ---------------------------------------------------------------------------
__global__ __launch_bounds__(512)
void proj_kernel(const float* __restrict__ x,
                 const float* __restrict__ wqk, const float* __restrict__ bqk,
                 const float* __restrict__ wv,  const float* __restrict__ bv,
                 _Float16* __restrict__ qk16, _Float16* __restrict__ vT16) {
  __shared__ _Float16 ldsW[2][128 * VPAD];   // [mat][row][40] padded

  const int tid  = threadIdx.x;
  const int lane = tid & 31;
  const int wave = tid >> 5;          // 0..15
  const int half = lane >> 4;
  const int lm   = lane & 15;
  const int mat  = wave >> 3;         // 0: qk, 1: v
  const int t0   = blockIdx.x * 128 + (wave & 7) * 16;

  const int   srow = tid >> 2;        // 0..127 staging row
  const int   scg  = tid & 3;         // 0..3 -> 8-wide column group
  const float* BB  = mat ? bv : bqk;

  v8f acc[8];
#pragma unroll
  for (int nt = 0; nt < 8; ++nt)
#pragma unroll
    for (int r = 0; r < 8; ++r) acc[nt][r] = 0.0f;

  const float* xrow = x + (size_t)(t0 + lm) * NDIM + half * 8;
  v16h a = load_ab_f32(xrow);                   // preload k-slice 0

  for (int k0 = 0; k0 < NDIM; k0 += 32) {
    __syncthreads();                            // previous slice consumed
    // ---- stage W[:,k0:k0+32] f32->f16 into LDS (both matrices) ----
#pragma unroll
    for (int m = 0; m < 2; ++m) {
      const float* wp = (m ? wv : wqk) + (size_t)srow * NDIM + k0 + scg * 8;
      v4f w0 = *(const v4f*)wp;
      v4f w1 = *(const v4f*)(wp + 4);
      v8h h;
#pragma unroll
      for (int i = 0; i < 4; ++i) {
        h[i]     = (_Float16)w0[i];
        h[i + 4] = (_Float16)w1[i];
      }
      *(v8h*)&ldsW[m][srow * VPAD + scg * 8] = h;
    }
    __syncthreads();

    // next x k-slice: global latency overlaps the WMMA chain below
    v16h anext = a;
    if (k0 + 32 < NDIM) anext = load_ab_f32(xrow + k0 + 32);

    // ---- 8 WMMA tiles, B operands 2-deep pipelined from LDS ----
    v16h bcur = load_ab_f16(&ldsW[mat][lm * VPAD + half * 8]);
#pragma unroll
    for (int nt = 0; nt < 8; ++nt) {
      v16h bnext = bcur;
      if (nt < 7)
        bnext = load_ab_f16(&ldsW[mat][((nt + 1) * 16 + lm) * VPAD + half * 8]);
      acc[nt] = wmma_f16(a, bcur, acc[nt]);
      bcur = bnext;
    }
    a = anext;
  }

  if (mat == 0) {
    // qk row-major: [token][128]
#pragma unroll
    for (int nt = 0; nt < 8; ++nt) {
      float bb = BB[nt * 16 + lm];
#pragma unroll
      for (int r = 0; r < 8; ++r) {
        qk16[(size_t)(t0 + r + 8 * half) * NDK + nt * 16 + lm] =
            (_Float16)(acc[nt][r] + bb);
      }
    }
  } else {
    // vT: [b][vdim][2048]; rows r=0..7 are consecutive tokens -> 16B store
    const int b  = t0 >> 11;
    const int n0 = t0 & (NSEQ - 1);
#pragma unroll
    for (int nt = 0; nt < 8; ++nt) {
      float bb = BB[nt * 16 + lm];
      v8h pack;
#pragma unroll
      for (int r = 0; r < 8; ++r) pack[r] = (_Float16)(acc[nt][r] + bb);
      _Float16* dst = vT16 + ((size_t)b * NDK + nt * 16 + lm) * NSEQ
                           + n0 + 8 * half;
      *(v8h*)dst = pack;
    }
  }
}

// ---------------------------------------------------------------------------
// Phase 2: fused flash-style attention.
// grid = 16 * 16 blocks of 256 threads (8 waves); wave owns 16 query rows.
// ---------------------------------------------------------------------------
__global__ __launch_bounds__(256)
void attn_kernel(const _Float16* __restrict__ qk16,
                 const _Float16* __restrict__ vT16,
                 float* __restrict__ out) {
  __shared__ _Float16 ldsK[2][32 * KPAD];
  __shared__ _Float16 ldsV[2][128 * VPAD];
  __shared__ _Float16 ldsP[8][16 * VPAD];

  const int tid  = threadIdx.x;
  const int lane = tid & 31;
  const int wave = tid >> 5;                       // 0..7
  const int half = lane >> 4;
  const int lm   = lane & 15;
  const int bidx = blockIdx.x >> 4;                // batch
  const int q0   = (blockIdx.x & 15) * 128 + wave * 16;

  _Float16* P = ldsP[wave];
  const _Float16* kbase = qk16 + (size_t)bidx * NSEQ * NDK;
  const _Float16* vbase = vT16 + (size_t)bidx * NDK * NSEQ;

  // Q tile: 4 A-operands (K = 128 = 4 x 32)
  v16h qa[4];
  const _Float16* qrow = kbase + (size_t)(q0 + lm) * NDK;
#pragma unroll
  for (int kc = 0; kc < 4; ++kc)
    qa[kc] = load_ab_f16(qrow + kc * 32 + half * 8);

  v8f o[8];
  float mr[8], lr[8];
#pragma unroll
  for (int vt = 0; vt < 8; ++vt)
#pragma unroll
    for (int r = 0; r < 8; ++r) o[vt][r] = 0.0f;
#pragma unroll
  for (int r = 0; r < 8; ++r) { mr[r] = -1e30f; lr[r] = 0.0f; }

  // ---- cooperative async staging: K tile 512 chunks, V tile 512 chunks ----
  auto stage = [&](int j0, int buf) {
#pragma unroll
    for (int s = 0; s < 2; ++s) {
      int c  = tid + s * 256;
      int kr = c >> 4, kcg = c & 15;               // K: 32 rows x 16 chunks
      async_b128(&ldsK[buf][kr * KPAD + kcg * 8],
                 kbase + (size_t)(j0 + kr) * NDK + kcg * 8);
      int vr = c >> 2, vcg = c & 3;                // V: 128 rows x 4 chunks
      async_b128(&ldsV[buf][vr * VPAD + vcg * 8],
                 vbase + (size_t)vr * NSEQ + j0 + vcg * 8);
    }
  };

  stage(0, 0);                                     // preload block 0
  for (int jb = 0; jb < NSEQ / 32; ++jb) {
    const int cur = jb & 1;
    if (jb + 1 < NSEQ / 32) {
      stage((jb + 1) * 32, cur ^ 1);               // overlap next block
      WAIT_ASYNC(4);                               // current buffer complete
    } else {
      WAIT_ASYNC(0);
    }
    __syncthreads();                               // all waves see K/V tiles

    // ---- S = Q (16x128) . K_block^T : two 16x16 C tiles, pipelined ----
    v8f cs0, cs1;
#pragma unroll
    for (int r = 0; r < 8; ++r) { cs0[r] = 0.0f; cs1[r] = 0.0f; }
    v16h kcur = load_ab_f16(&ldsK[cur][lm * KPAD + half * 8]);
#pragma unroll
    for (int i = 0; i < 8; ++i) {                  // i = kc*2 + tile
      v16h knext = kcur;
      if (i < 7) {
        const int kc = (i + 1) >> 1, t = (i + 1) & 1;
        knext = load_ab_f16(&ldsK[cur][(t * 16 + lm) * KPAD
                                       + kc * 32 + half * 8]);
      }
      if (i & 1) cs1 = wmma_f16(qa[i >> 1], kcur, cs1);
      else       cs0 = wmma_f16(qa[i >> 1], kcur, cs0);
      kcur = knext;
    }

    // ---- online softmax: row stats via 16-lane butterfly reductions ----
    float scale[8];
#pragma unroll
    for (int r = 0; r < 8; ++r) {
      float mx = fmaxf(cs0[r], cs1[r]);
      mx = fmaxf(mx, __shfl_xor(mx, 1, 32));
      mx = fmaxf(mx, __shfl_xor(mx, 2, 32));
      mx = fmaxf(mx, __shfl_xor(mx, 4, 32));
      mx = fmaxf(mx, __shfl_xor(mx, 8, 32));
      float mn = fmaxf(mr[r], mx);
      float sc = __expf(mr[r] - mn);
      float p0 = __expf(cs0[r] - mn);
      float p1 = __expf(cs1[r] - mn);
      P[(r + 8 * half) * VPAD + lm]      = (_Float16)p0;
      P[(r + 8 * half) * VPAD + 16 + lm] = (_Float16)p1;
      float s = p0 + p1;
      s += __shfl_xor(s, 1, 32);
      s += __shfl_xor(s, 2, 32);
      s += __shfl_xor(s, 4, 32);
      s += __shfl_xor(s, 8, 32);
      mr[r]    = mn;
      lr[r]    = lr[r] * sc + s;
      scale[r] = sc;
    }
#pragma unroll
    for (int vt = 0; vt < 8; ++vt)
#pragma unroll
      for (int r = 0; r < 8; ++r) o[vt][r] *= scale[r];

    // P region is wave-private; LDS ops of a wave complete in order, so no
    // block barrier is needed between the P stores above and this read.
    v16h pa = load_ab_f16(&P[lm * VPAD + half * 8]);

    // ---- O += P (16x32) . V_block (32x128), V operands pipelined ----
    v16h vcur = load_ab_f16(&ldsV[cur][lm * VPAD + half * 8]);
#pragma unroll
    for (int vt = 0; vt < 8; ++vt) {
      v16h vnext = vcur;
      if (vt < 7)
        vnext = load_ab_f16(&ldsV[cur][((vt + 1) * 16 + lm) * VPAD + half * 8]);
      o[vt] = wmma_f16(pa, vcur, o[vt]);
      vcur = vnext;
    }
    __syncthreads();      // reads of buffer `cur` done before it is restaged
  }

  // ---- finalize: O / l, store f32 out[b][token][vdim] ----
  float inv[8];
#pragma unroll
  for (int r = 0; r < 8; ++r) inv[r] = 1.0f / lr[r];
#pragma unroll
  for (int vt = 0; vt < 8; ++vt)
#pragma unroll
    for (int r = 0; r < 8; ++r) {
      out[((size_t)bidx * NSEQ + q0 + r + 8 * half) * NDK + vt * 16 + lm] =
          o[vt][r] * inv[r];
    }
}

// ---------------------------------------------------------------------------
extern "C" void kernel_launch(void* const* d_in, const int* in_sizes, int n_in,
                              void* d_out, int out_size, void* d_ws, size_t ws_size,
                              hipStream_t stream) {
  (void)in_sizes; (void)n_in; (void)out_size; (void)ws_size;
  const float* x     = (const float*)d_in[0];
  const float* wqk_w = (const float*)d_in[1];
  const float* wqk_b = (const float*)d_in[2];
  const float* wv_w  = (const float*)d_in[3];
  const float* wv_b  = (const float*)d_in[4];
  float* out = (float*)d_out;

  _Float16* qk16 = (_Float16*)d_ws;                     // 8 MB
  _Float16* vT16 = qk16 + (size_t)NB * NSEQ * NDK;      // 8 MB

  proj_kernel<<<(NB * NSEQ) / 128, 512, 0, stream>>>(x, wqk_w, wqk_b,
                                                     wv_w, wv_b, qk16, vT16);
  attn_kernel<<<NB * (NSEQ / 128), 256, 0, stream>>>(qk16, vT16, out);
}
